// CNNModel_82222853915196
// MI455X (gfx1250) — matrix operand order repaired and checked by reading, not compiled, not run on hardware
//
#include <hip/hip_runtime.h>
#include <hip/hip_bf16.h>

typedef __attribute__((ext_vector_type(16))) _Float16 v16h;
typedef __attribute__((ext_vector_type(8)))  _Float16 v8h;
typedef __attribute__((ext_vector_type(8)))  float    v8f;
typedef __attribute__((ext_vector_type(4)))  unsigned int v4u;
typedef __attribute__((ext_vector_type(8)))  int      v8i;
typedef __attribute__((ext_vector_type(4)))  int      v4i;

#define BATCH 32
#define FREQ 64
#define TT 512
#define OUTD 400
#define FEATS 8192           // 16*T
#define FCIN 8256            // 64 + 8192

// ---------------------------------------------------------------------------
// Prep kernel A: Wfc (400 x 8256 f32) -> f16 copy (read once, used by ~25600
// tiles; halves the dominant weight traffic and enables b128 fragment loads).
// ---------------------------------------------------------------------------
__global__ __launch_bounds__(256) void conv_wfc_kernel(
    const float* __restrict__ Wfc, _Float16* __restrict__ wfc16)
{
    const int i = blockIdx.x * 256 + threadIdx.x;
    if (i < OUTD * FCIN) wfc16[i] = (_Float16)Wfc[i];
}

// ---------------------------------------------------------------------------
// Prep kernel B: batch (B,F,T) f32 -> xT16 (B,T,F) f16, so the frame-GEMM
// A-fragments become contiguous 16-byte loads per lane.
// ---------------------------------------------------------------------------
__global__ __launch_bounds__(256) void transpose_x_kernel(
    const float* __restrict__ batch, _Float16* __restrict__ xT16)
{
    const int i = blockIdx.x * 256 + threadIdx.x;      // over 32*512*64
    const int b = i >> 15;
    const int r = i & 32767;
    const int t = r >> 6;
    const int f = r & 63;
    xT16[i] = (_Float16)batch[(b << 15) + (f << 9) + t];
}

// ---------------------------------------------------------------------------
// Kernel 1: feature extraction. One block per utterance; whole utterance
// (64*512 f32 = 128 KB) staged into dynamic LDS via the Tensor Data Mover
// (2D tile 512 dwords x 64 rows), then 256 threads compute the 8192 features
// (tiny FLOP count) and emit them in f16 for the WMMA consumer.
// ---------------------------------------------------------------------------
__global__ __launch_bounds__(256) void feat_kernel(
    const float* __restrict__ batch,
    const float* __restrict__ W8,  const float* __restrict__ b8,
    const float* __restrict__ W16, const float* __restrict__ b16,
    const float* __restrict__ W32, const float* __restrict__ b32,
    const float* __restrict__ W64, const float* __restrict__ b64,
    _Float16* __restrict__ feats16)
{
    extern __shared__ float xs[];                   // 64*512 floats = 128 KB
    const int b = blockIdx.x;

#if __has_builtin(__builtin_amdgcn_tensor_load_to_lds)
    // TDM path: wave 0 issues one DMA descriptor for the whole 128 KB tile.
    if ((threadIdx.x >> 5) == 0) {
        const unsigned long long ga =
            (unsigned long long)(const void*)(batch + (size_t)b * FREQ * TT);
        // D# group0: count=1, lds_addr=0, global_addr, type=2 ("image")
        v4u g0 = { 1u,                                  // count=1, user desc
                   0u,                                  // lds_addr = 0
                   (unsigned int)(ga & 0xFFFFFFFFu),
                   (unsigned int)((ga >> 32) & 0x1FFFFFFu) | (2u << 30) };
        // D# group1: data_size=4B; tensor_dim0=512, tensor_dim1=64,
        // tile_dim0=512, tile_dim1=64, dim0_stride=512, dim1_stride=32768
        v8i g1 = { (int)0x00020000,        // data_size=2 (4B), mask=0
                   (int)0x02000000,        // tensor_dim0 lo16 (=512) << 16
                   (int)0x00400000,        // tensor_dim1 (=64) << 16
                   (int)0x02000000,        // tile_dim0 (=512) << 16
                   (int)0x00000040,        // tile_dim1 = 64
                   (int)0x00000200,        // tensor_dim0_stride = 512
                   (int)0x80000000,        // tensor_dim1_stride lo16 << 16
                   (int)0x00000000 };
        v4i g2 = { 0, 0, 0, 0 };
        v4i g3 = { 0, 0, 0, 0 };
        v8i g4 = { 0, 0, 0, 0, 0, 0, 0, 0 };
        __builtin_amdgcn_tensor_load_to_lds(g0, g1, g2, g3, g4, 0);
        __builtin_amdgcn_s_wait_tensorcnt(0);
    }
    __syncthreads();
#else
    const float* xb = batch + (size_t)b * FREQ * TT;
    for (int i = threadIdx.x; i < FREQ * TT; i += 256) xs[i] = xb[i];
    __syncthreads();
#endif

    _Float16* fb = feats16 + (size_t)b * FEATS;
    const int tid = threadIdx.x;

    // f8: (8 shifts, 64 patches, 4 out), K = 64
    for (int e = tid; e < 2048; e += 256) {
        const int k = e >> 8, rem = e & 255, p = rem >> 2, o = rem & 3;
        float acc = b8[k * 4 + o];
        const float* w  = W8 + k * 256 + o * 64;
        const float* xr = xs + k * TT + p * 8;
        for (int i = 0; i < 8; ++i)
            #pragma unroll
            for (int j = 0; j < 8; ++j)
                acc += xr[i * TT + j] * w[i * 8 + j];
        fb[e] = (_Float16)acc;
    }
    // f16: (4 shifts, 32 patches, 16 out), K = 256
    for (int e = tid; e < 2048; e += 256) {
        const int k = e >> 9, rem = e & 511, p = rem >> 4, o = rem & 15;
        float acc = b16[k * 16 + o];
        const float* w  = W16 + k * 4096 + o * 256;
        const float* xr = xs + k * TT + p * 16;
        for (int i = 0; i < 16; ++i)
            #pragma unroll
            for (int j = 0; j < 16; ++j)
                acc += xr[i * TT + j] * w[i * 16 + j];
        fb[2048 + e] = (_Float16)acc;
    }
    // f32: (2 shifts, 16 patches, 64 out), K = 1024
    for (int e = tid; e < 2048; e += 256) {
        const int k = e >> 10, rem = e & 1023, p = rem >> 6, o = rem & 63;
        float acc = b32[k * 64 + o];
        const float* w  = W32 + k * 65536 + o * 1024;
        const float* xr = xs + k * TT + p * 32;
        for (int i = 0; i < 32; ++i)
            #pragma unroll
            for (int j = 0; j < 32; ++j)
                acc += xr[i * TT + j] * w[i * 32 + j];
        fb[4096 + e] = (_Float16)acc;
    }
    // f64: (8 patches, 256 out), K = 4096
    for (int e = tid; e < 2048; e += 256) {
        const int p = e >> 8, o = e & 255;
        float acc = b64[o];
        const float* w  = W64 + (size_t)o * 4096;
        const float* xr = xs + p * 64;
        for (int i = 0; i < 64; ++i)
            #pragma unroll
            for (int j = 0; j < 64; ++j)
                acc += xr[i * TT + j] * w[i * 64 + j];
        fb[6144 + e] = (_Float16)acc;
    }
}

// ---------------------------------------------------------------------------
// Kernel 2: c[b][o] = bfc[o] + feats[b] . Wfc[o, 64:]   (M=32, N=400, K=8192)
// One wave per 16x16 tile; fragments loaded as aligned v8h/v16h vectors.
// ---------------------------------------------------------------------------
__global__ __launch_bounds__(32) void featfc_kernel(
    const _Float16* __restrict__ feats16,
    const _Float16* __restrict__ wfc16,
    const float* __restrict__ bfc,
    float* __restrict__ cws)
{
    const int Mtile = blockIdx.x;       // 0..1 (batch groups of 16)
    const int Ntile = blockIdx.y;       // 0..24 (output cols)
    const int L    = threadIdx.x;
    const int lmod = L & 15;
    const int lhi  = L >> 4;

    const int bRow = Mtile * 16 + lmod;
    const int oCol = Ntile * 16 + lmod;
    const _Float16* fA = feats16 + (size_t)bRow * FEATS;
    const _Float16* wB = wfc16 + (size_t)oCol * FCIN + 64;

    v8f acc = {};
    #pragma unroll 2
    for (int k0 = 0; k0 < FEATS; k0 += 32) {
        // A (16x32 f16): lanes 0-15 K=0..15, lanes 16-31 K=16..31
        v8h a0 = *(const v8h*)(fA + k0 + lhi * 16);
        v8h a1 = *(const v8h*)(fA + k0 + 8 + lhi * 16);
        v16h a = __builtin_shufflevector(a0, a1, 0,1,2,3,4,5,6,7,
                                                 8,9,10,11,12,13,14,15);
        // B (32x16 f16): lanes 0-15 K=0..15, lanes 16-31 K=16..31
        v16h bb = *(const v16h*)(wB + k0 + lhi * 16);
        acc = __builtin_amdgcn_wmma_f32_16x16x32_f16(
            false, a, false, bb, (short)0, acc, false, false);
    }
    #pragma unroll
    for (int r = 0; r < 8; ++r) {
        const int bIdx = Mtile * 16 + r + (lhi ? 8 : 0);
        cws[bIdx * OUTD + oCol] = acc[r] + bfc[oCol];
    }
}

// ---------------------------------------------------------------------------
// Kernel 3: out[b*T+t][o] = c[b][o] + sum_f x[b][f][t] * Wfc[o][f]
// M=16384, N=400, K=64 (2 WMMA steps); accumulator seeded with c[b][o].
// ---------------------------------------------------------------------------
__global__ __launch_bounds__(32) void frame_kernel(
    const _Float16* __restrict__ xT16,       // [b][t][f] f16
    const _Float16* __restrict__ wfc16,
    const float* __restrict__ cws,
    float* __restrict__ out)
{
    const int Mtile = blockIdx.x;       // 0..1023
    const int Ntile = blockIdx.y;       // 0..24
    const int L    = threadIdx.x;
    const int lmod = L & 15;
    const int lhi  = L >> 4;

    const int b  = Mtile >> 5;
    const int t  = ((Mtile & 31) << 4) + lmod;
    const int oCol = Ntile * 16 + lmod;

    const _Float16* xA = xT16 + ((size_t)(b * TT + t)) * FREQ;
    const _Float16* wB = wfc16 + (size_t)oCol * FCIN;

    const float cval = cws[b * OUTD + oCol];
    v8f acc;
    #pragma unroll
    for (int r = 0; r < 8; ++r) acc[r] = cval;

    #pragma unroll
    for (int k0 = 0; k0 < 64; k0 += 32) {
        v8h a0 = *(const v8h*)(xA + k0 + lhi * 16);
        v8h a1 = *(const v8h*)(xA + k0 + 8 + lhi * 16);
        v16h a = __builtin_shufflevector(a0, a1, 0,1,2,3,4,5,6,7,
                                                 8,9,10,11,12,13,14,15);
        v16h bb = *(const v16h*)(wB + k0 + lhi * 16);
        acc = __builtin_amdgcn_wmma_f32_16x16x32_f16(
            false, a, false, bb, (short)0, acc, false, false);
    }
    #pragma unroll
    for (int r = 0; r < 8; ++r) {
        const int row = Mtile * 16 + r + (lhi ? 8 : 0);
        out[(size_t)row * OUTD + oCol] = acc[r];
    }
}

// ---------------------------------------------------------------------------
extern "C" void kernel_launch(void* const* d_in, const int* in_sizes, int n_in,
                              void* d_out, int out_size, void* d_ws, size_t ws_size,
                              hipStream_t stream)
{
    (void)in_sizes; (void)n_in; (void)out_size; (void)ws_size;
    const float* batch = (const float*)d_in[0];
    const float* W8    = (const float*)d_in[1];
    const float* b8    = (const float*)d_in[2];
    const float* W16   = (const float*)d_in[3];
    const float* b16   = (const float*)d_in[4];
    const float* W32   = (const float*)d_in[5];
    const float* b32   = (const float*)d_in[6];
    const float* W64   = (const float*)d_in[7];
    const float* b64   = (const float*)d_in[8];
    const float* Wfc   = (const float*)d_in[9];
    const float* bfc   = (const float*)d_in[10];
    // d_in[11] = lengths (int64): reference assumes full-length utterances.

    // Workspace layout (all offsets 32-byte aligned):
    char* ws = (char*)d_ws;
    float*    cws     = (float*)ws;                                  //  51200 B
    _Float16* wfc16   = (_Float16*)(ws + 51200);                     // 6604800 B
    _Float16* xT16    = (_Float16*)(ws + 51200 + 6604800);           // 2097152 B
    _Float16* feats16 = (_Float16*)(ws + 51200 + 6604800 + 2097152); //  524288 B
    float* outp = (float*)d_out;

    conv_wfc_kernel<<<(OUTD * FCIN + 255) / 256, 256, 0, stream>>>(Wfc, wfc16);
    transpose_x_kernel<<<(BATCH * TT * FREQ) / 256, 256, 0, stream>>>(batch, xT16);

    feat_kernel<<<BATCH, 256, FREQ * TT * sizeof(float), stream>>>(
        batch, W8, b8, W16, b16, W32, b32, W64, b64, feats16);

    featfc_kernel<<<dim3(2, 25), 32, 0, stream>>>(feats16, wfc16, bfc, cws);

    frame_kernel<<<dim3(1024, 25), 32, 0, stream>>>(xT16, wfc16, cws, outp);
}
